// CDAG_86423331930377
// MI455X (gfx1250) — matrix operand order repaired and checked by reading, not compiled, not run on hardware
//
#include <hip/hip_runtime.h>
#include <math.h>

typedef __attribute__((ext_vector_type(16))) _Float16 v16h;
typedef __attribute__((ext_vector_type(8)))  _Float16 v8h;
typedef __attribute__((ext_vector_type(8)))  float    v8f;
typedef __attribute__((ext_vector_type(4)))  unsigned u32x4;
typedef __attribute__((ext_vector_type(8)))  int      i32x8;
typedef __attribute__((ext_vector_type(4)))  int      i32x4;

#define NPTS 600000

// ---------------------------------------------------------------------------
// Pack f32 weights [K][cin][cout] -> f16 flat buffer in the LDS B-operand
// layout: dst[(((k*chunks+ch)*2 + t)*16 + nn)*32 + ci]  (cout = t*16+nn,
// cin = ch*32+ci). This buffer is then TDM-DMA'd into LDS by each workgroup.
// ---------------------------------------------------------------------------
__global__ void pack_weights_kernel(const float* __restrict__ W,
                                    _Float16* __restrict__ dst,
                                    int K, int chunks) {
    const int cinTot = chunks << 5;
    const int wElems = K * cinTot * 32;
    for (int i = blockIdx.x * blockDim.x + threadIdx.x; i < wElems;
         i += gridDim.x * blockDim.x) {
        int k    = i / (cinTot * 32);
        int r    = i - k * (cinTot * 32);
        int cing = r >> 5;
        int cout = r & 31;
        int ch   = cing >> 5;
        int ci   = cing & 31;
        int t    = cout >> 4;
        int nn   = cout & 15;
        dst[((((k * chunks + ch) * 2 + t) * 16 + nn) << 5) + ci] = (_Float16)W[i];
    }
}

// ---------------------------------------------------------------------------
// Generic WMMA sparse conv:  out[n, 0:32] = sum_k  in[nbr[k,n], :] @ W[k]
// Weights (pre-packed f16) are DMA'd global->LDS with the Tensor Data Mover.
//
// A layout (16-bit A 16x32, contiguous K): lane L, g=L>>4, row=L&15 holds
//   K = [16g..16g+7] in halfs 0..7 and K=[16g+8..16g+15] in halfs 8..15.
// B layout (32x16): lane L holds column n=row, K=[16g..16g+15] contiguous.
// C/D layout: vgpr r, lane L -> M = r + 8g, N = row.
// ---------------------------------------------------------------------------
template <int K, int CHUNKS>
__global__ void __launch_bounds__(256) conv_wmma_kernel(
    const _Float16* __restrict__ in,    // [n, 32*CHUNKS] f16
    const int*      __restrict__ nbr,   // [K, n] or nullptr (identity)
    const _Float16* __restrict__ Wf16,  // packed weights, WELEMS halfs
    float*          __restrict__ out,   // [n, 32] f32
    float*          __restrict__ stats, // 128-float stat block or nullptr
    const float*    __restrict__ bias,  // [32] or nullptr
    int n)
{
    extern __shared__ _Float16 ldsW[];
    constexpr int WELEMS = K * CHUNKS * 32 * 32;

    // --- TDM: one wave issues a 1-D tensor_load_to_lds of the packed weights ---
    if ((threadIdx.x >> 5) == 0) {
        unsigned ldsbase = __builtin_amdgcn_groupstaticsize();
        unsigned long long ga = (unsigned long long)Wf16;
        const unsigned E = (unsigned)WELEMS;   // f16 elements (tile_dim0 < 2^16)
        u32x4 g0;
        g0[0] = 1u;                                        // count=1, user mode
        g0[1] = ldsbase;                                   // lds_addr
        g0[2] = (unsigned)(ga & 0xffffffffull);            // global_addr[31:0]
        g0[3] = (unsigned)((ga >> 32) & 0x1ffffffull) | (2u << 30); // ga[56:32] | type=2
        i32x8 g1;
        g1[0] = (int)(1u << 16);                 // workgroup_mask=0, data_size=1 (2B)
        g1[1] = (int)((E & 0xffffu) << 16);      // tensor_dim0[15:0]
        g1[2] = (int)((E >> 16) | (1u << 16));   // tensor_dim0[31:16], tensor_dim1=1
        g1[3] = (int)((E & 0xffffu) << 16);      // tile_dim0 = E
        g1[4] = 1;                               // tile_dim1 = 1, tile_dim2 = 0
        g1[5] = (int)E;                          // tensor_dim0_stride[31:0]
        g1[6] = 0;
        g1[7] = 0;
        i32x4 g2 = {0, 0, 0, 0};
        i32x4 g3 = {0, 0, 0, 0};
#if __clang_major__ >= 23
        i32x8 g4 = {0, 0, 0, 0, 0, 0, 0, 0};
        __builtin_amdgcn_tensor_load_to_lds(g0, g1, g2, g3, g4, 0);
#else
        __builtin_amdgcn_tensor_load_to_lds(g0, g1, g2, g3, 0);
#endif
        __builtin_amdgcn_s_wait_tensorcnt((short)0);
    }
    __syncthreads();

    constexpr int cinTot = CHUNKS << 5;
    const int lane = threadIdx.x & 31;
    const int row  = lane & 15;
    const int g    = lane >> 4;
    const int wavesPerBlk = blockDim.x >> 5;
    const int wid = blockIdx.x * wavesPerBlk + (threadIdx.x >> 5);
    const int nw  = gridDim.x * wavesPerBlk;
    const int ntiles = n >> 4;

    float s1a = 0.f, s2a = 0.f, s1b = 0.f, s2b = 0.f;
    float ba = 0.f, bb = 0.f;
    if (bias) { ba = bias[row]; bb = bias[16 + row]; }

    for (int tile = wid; tile < ntiles; tile += nw) {
        const int base = tile << 4;
        v8f acc0 = {};
        v8f acc1 = {};
#pragma unroll 3
        for (int k = 0; k < K; ++k) {
            const int idx = nbr ? nbr[(size_t)k * n + base + row] : (base + row);
            const _Float16* ap0 = in + (size_t)idx * cinTot;
#pragma unroll
            for (int ch = 0; ch < CHUNKS; ++ch) {
                const _Float16* ap = ap0 + (ch << 5);
                v8h alo = *(const v8h*)(ap + (g << 4));
                v8h ahi = *(const v8h*)(ap + 8 + (g << 4));
                v16h a;
#pragma unroll
                for (int j = 0; j < 8; ++j) { a[j] = alo[j]; a[8 + j] = ahi[j]; }
                const _Float16* bp0 =
                    ldsW + ((((((k * CHUNKS + ch) * 2 + 0) * 16 + row)) << 5) + (g << 4));
                const _Float16* bp1 =
                    ldsW + ((((((k * CHUNKS + ch) * 2 + 1) * 16 + row)) << 5) + (g << 4));
                v16h b0 = *(const v16h*)bp0;
                v16h b1 = *(const v16h*)bp1;
                acc0 = __builtin_amdgcn_wmma_f32_16x16x32_f16(
                    false, a, false, b0, (short)0, acc0, false, false);
                acc1 = __builtin_amdgcn_wmma_f32_16x16x32_f16(
                    false, a, false, b1, (short)0, acc1, false, false);
            }
        }
#pragma unroll
        for (int r = 0; r < 8; ++r) {
            float v0 = acc0[r] + ba;
            float v1 = acc1[r] + bb;
            float* orow = out + ((size_t)(base + (g << 3) + r) << 5);
            orow[row]      = v0;
            orow[16 + row] = v1;
            s1a += v0; s2a += v0 * v0;
            s1b += v1; s2b += v1 * v1;
        }
    }
    if (stats) {
        atomicAdd(&stats[row],           s1a);
        atomicAdd(&stats[32 + row],      s2a);
        atomicAdd(&stats[16 + row],      s1b);
        atomicAdd(&stats[32 + 16 + row], s2b);
    }
}

// --------------------------- small helper kernels ---------------------------

__global__ void zero_small_kernel(float* p, int n) {
    int i = blockIdx.x * blockDim.x + threadIdx.x;
    if (i < n) p[i] = 0.f;
}

// x -> f16 copy, and into cat low half
__global__ void prep_kernel(const float* __restrict__ x, _Float16* __restrict__ xf16,
                            _Float16* __restrict__ cat, int n) {
    int i = blockIdx.x * blockDim.x + threadIdx.x;
    if (i < n * 32) {
        float v = x[i];
        xf16[i] = (_Float16)v;
        int p = i >> 5, c = i & 31;
        cat[((size_t)p << 6) + c] = (_Float16)v;
    }
}

// stats (sum, sum^2) of yg = g@wg and yx = x@wx  (biases cancel in BN)
__global__ void __launch_bounds__(256) stage1_stats_kernel(
    const float* __restrict__ x, const float* __restrict__ gfeat,
    const float* __restrict__ wgw, const float* __restrict__ wxw,
    float* st_g, float* st_x, int n) {
    __shared__ float red[4][16];
    for (int i = threadIdx.x; i < 64; i += blockDim.x) ((float*)red)[i] = 0.f;
    __syncthreads();
    float sg1[16] = {}, sg2[16] = {}, sx1[16] = {}, sx2[16] = {};
    for (int p = blockIdx.x * blockDim.x + threadIdx.x; p < n;
         p += gridDim.x * blockDim.x) {
        const float* xr = x + (size_t)p * 32;
        const float* gr = gfeat + (size_t)p * 32;
        for (int co = 0; co < 16; ++co) {
            float yg = 0.f, yx = 0.f;
            for (int ci = 0; ci < 32; ++ci) {
                yg += gr[ci] * wgw[ci * 16 + co];
                yx += xr[ci] * wxw[ci * 16 + co];
            }
            sg1[co] += yg; sg2[co] += yg * yg;
            sx1[co] += yx; sx2[co] += yx * yx;
        }
    }
    for (int co = 0; co < 16; ++co) {
        atomicAdd(&red[0][co], sg1[co]); atomicAdd(&red[1][co], sg2[co]);
        atomicAdd(&red[2][co], sx1[co]); atomicAdd(&red[3][co], sx2[co]);
    }
    __syncthreads();
    if (threadIdx.x < 16) {
        int c = threadIdx.x;
        atomicAdd(&st_g[c],      red[0][c]); atomicAdd(&st_g[32 + c], red[1][c]);
        atomicAdd(&st_x[c],      red[2][c]); atomicAdd(&st_x[32 + c], red[3][c]);
    }
}

// st: [0:32]=sum [32:64]=sum^2 -> write [64:96]=mean [96:128]=inv_std
__global__ void finalize_stats_kernel(float* st, int nch, float cnt) {
    int c = threadIdx.x;
    if (c < nch) {
        float m = st[c] / cnt;
        float v = st[32 + c] / cnt - m * m;
        st[64 + c] = m;
        st[96 + c] = rsqrtf(fmaxf(v, 0.f) + 1e-5f);
    }
}

// s = relu(BN(yg)+BN(yx)) @ psi_w ; store raw s + stats(1 ch)
__global__ void __launch_bounds__(256) stage3_kernel(
    const float* __restrict__ x, const float* __restrict__ gfeat,
    const float* __restrict__ wgw, const float* __restrict__ wxw,
    const float* __restrict__ psiw, const float* __restrict__ st_g,
    const float* __restrict__ st_x, float* __restrict__ s_raw,
    float* st_s, int n) {
    __shared__ float red[2];
    if (threadIdx.x < 2) red[threadIdx.x] = 0.f;
    __syncthreads();
    float a1 = 0.f, a2 = 0.f;
    for (int p = blockIdx.x * blockDim.x + threadIdx.x; p < n;
         p += gridDim.x * blockDim.x) {
        const float* xr = x + (size_t)p * 32;
        const float* gr = gfeat + (size_t)p * 32;
        float s = 0.f;
        for (int co = 0; co < 16; ++co) {
            float yg = 0.f, yx = 0.f;
            for (int ci = 0; ci < 32; ++ci) {
                yg += gr[ci] * wgw[ci * 16 + co];
                yx += xr[ci] * wxw[ci * 16 + co];
            }
            float ng = (yg - st_g[64 + co]) * st_g[96 + co];
            float nx = (yx - st_x[64 + co]) * st_x[96 + co];
            float ps = ng + nx;
            ps = ps > 0.f ? ps : 0.f;
            s += ps * psiw[co];
        }
        s_raw[p] = s;
        a1 += s; a2 += s * s;
    }
    atomicAdd(&red[0], a1); atomicAdd(&red[1], a2);
    __syncthreads();
    if (threadIdx.x == 0) { atomicAdd(&st_s[0], red[0]); atomicAdd(&st_s[32], red[1]); }
}

__global__ void asp_kernel(const float* __restrict__ s_raw, const float* __restrict__ st_s,
                           float* __restrict__ A_sp, int n) {
    int i = blockIdx.x * blockDim.x + threadIdx.x;
    if (i < n) {
        float z = (s_raw[i] - st_s[64]) * st_s[96];
        A_sp[i] = 1.f / (1.f + expf(-z));
    }
}

__device__ __forceinline__ unsigned enc_f32(float v) {
    unsigned b = __float_as_uint(v);
    return (b & 0x80000000u) ? ~b : (b | 0x80000000u);
}

// per-batch segment sum / max of x
__global__ void __launch_bounds__(256) seg_kernel(
    const float* __restrict__ x, const int* __restrict__ bidx,
    float* g_cnt, float* g_sum, unsigned* g_max, int n) {
    __shared__ float s_sum[256];
    __shared__ unsigned s_max[256];
    __shared__ float s_cnt[8];
    for (int i = threadIdx.x; i < 256; i += blockDim.x) { s_sum[i] = 0.f; s_max[i] = 0u; }
    if (threadIdx.x < 8) s_cnt[threadIdx.x] = 0.f;
    __syncthreads();
    for (int p = blockIdx.x * blockDim.x + threadIdx.x; p < n;
         p += gridDim.x * blockDim.x) {
        int b = bidx[p];
        const float* xr = x + (size_t)p * 32;
        atomicAdd(&s_cnt[b], 1.f);
        for (int c = 0; c < 32; ++c) {
            float v = xr[c];
            atomicAdd(&s_sum[b * 32 + c], v);
            atomicMax(&s_max[b * 32 + c], enc_f32(v));
        }
    }
    __syncthreads();
    for (int i = threadIdx.x; i < 256; i += blockDim.x) {
        atomicAdd(&g_sum[i], s_sum[i]);
        atomicMax(&g_max[i], s_max[i]);
    }
    if (threadIdx.x < 8) atomicAdd(&g_cnt[threadIdx.x], s_cnt[threadIdx.x]);
}

// A_ch[b][c] = sigmoid(mlp(avg)) + sigmoid(mlp(max))   (8x32, 1 block)
__global__ void chatt_kernel(const float* __restrict__ cnt, const float* __restrict__ sum,
                             const unsigned* __restrict__ mx,
                             const float* __restrict__ fc1w, const float* __restrict__ fc1b,
                             const float* __restrict__ fc2w, const float* __restrict__ fc2b,
                             float* __restrict__ a_ch) {
    int t = threadIdx.x;
    if (t >= 256) return;
    int b = t >> 5, c = t & 31;
    float cc = fmaxf(cnt[b], 1.f);
    float o_total = 0.f;
    for (int which = 0; which < 2; ++which) {
        float h[8];
        for (int j = 0; j < 8; ++j) {
            float z = fc1b[j];
            for (int ci = 0; ci < 32; ++ci) {
                float v;
                if (which == 0) v = sum[b * 32 + ci] / cc;
                else {
                    unsigned e = mx[b * 32 + ci];
                    unsigned bits = (e & 0x80000000u) ? (e ^ 0x80000000u) : ~e;
                    v = __uint_as_float(bits);
                }
                z += v * fc1w[ci * 8 + j];
            }
            h[j] = z > 0.f ? z : 0.f;
        }
        float o = fc2b[c];
        for (int j = 0; j < 8; ++j) o += h[j] * fc2w[j * 32 + c];
        o_total += 1.f / (1.f + expf(-o));
    }
    a_ch[t] = o_total;
}

// xi = relu(BN(raw)); write f16 copy and f32 fused init
__global__ void xi_kernel(const float* __restrict__ raw, const float* __restrict__ st,
                          _Float16* __restrict__ xi16, float* __restrict__ fused, int n32) {
    int i = blockIdx.x * blockDim.x + threadIdx.x;
    if (i < n32) {
        int c = i & 31;
        float v = (raw[i] - st[64 + c]) * st[96 + c];
        v = v > 0.f ? v : 0.f;
        xi16[i] = (_Float16)v;
        fused[i] = v;
    }
}

// fused += relu(BN(raw))
__global__ void fuse_kernel(const float* __restrict__ raw, const float* __restrict__ st,
                            float* __restrict__ fused, int n32) {
    int i = blockIdx.x * blockDim.x + threadIdx.x;
    if (i < n32) {
        int c = i & 31;
        float v = (raw[i] - st[64 + c]) * st[96 + c];
        v = v > 0.f ? v : 0.f;
        fused[i] += v;
    }
}

// A_ms = sigmoid(fused @ ms_out_w + ms_out_b)
__global__ void ams_kernel(const float* __restrict__ fused, const float* __restrict__ mow,
                           const float* __restrict__ mob, float* __restrict__ A_ms, int n) {
    int p = blockIdx.x * blockDim.x + threadIdx.x;
    if (p < n) {
        const float* fr = fused + (size_t)p * 32;
        float s = mob[0];
        for (int c = 0; c < 32; ++c) s += fr[c] * mow[c];
        A_ms[p] = 1.f / (1.f + expf(-s));
    }
}

// t = ((A_sp + A_ms) * A_ch[bidx]) * x  -> f16
__global__ void pattn1pre_kernel(const float* __restrict__ x, const float* __restrict__ A_sp,
                                 const float* __restrict__ A_ms, const float* __restrict__ a_ch,
                                 const int* __restrict__ bidx, _Float16* __restrict__ t16,
                                 int n32) {
    int i = blockIdx.x * blockDim.x + threadIdx.x;
    if (i < n32) {
        int p = i >> 5, c = i & 31;
        float v = (A_sp[p] + A_ms[p]) * a_ch[bidx[p] * 32 + c] * x[i];
        t16[i] = (_Float16)v;
    }
}

// cat high half = f16(pattn1 raw, bias already added by conv)
__global__ void cat_hi_kernel(const float* __restrict__ raw, _Float16* __restrict__ cat,
                              int n32) {
    int i = blockIdx.x * blockDim.x + threadIdx.x;
    if (i < n32) {
        int p = i >> 5, c = i & 31;
        cat[((size_t)p << 6) + 32 + c] = (_Float16)raw[i];
    }
}

// result = x + sigmoid(BN(px_raw)) * (A_sp * x) -> f16
__global__ void final_pre_kernel(const float* __restrict__ x, const float* __restrict__ rawpx,
                                 const float* __restrict__ st, const float* __restrict__ A_sp,
                                 _Float16* __restrict__ res16, int n32) {
    int i = blockIdx.x * blockDim.x + threadIdx.x;
    if (i < n32) {
        int p = i >> 5, c = i & 31;
        float z = (rawpx[i] - st[64 + c]) * st[96 + c];
        float pa = 1.f / (1.f + expf(-z));
        float xv = x[i];
        res16[i] = (_Float16)(xv + pa * (A_sp[p] * xv));
    }
}

// ---------------------------------------------------------------------------

extern "C" void kernel_launch(void* const* d_in, const int* in_sizes, int n_in,
                              void* d_out, int out_size, void* d_ws, size_t ws_size,
                              hipStream_t stream) {
    const int n = NPTS;
    const float* x     = (const float*)d_in[0];
    const float* gfeat = (const float*)d_in[1];
    const float* wgw   = (const float*)d_in[2];
    const float* wxw   = (const float*)d_in[4];
    const float* psiw  = (const float*)d_in[6];
    const float* fc1w  = (const float*)d_in[8];
    const float* fc1b  = (const float*)d_in[9];
    const float* fc2w  = (const float*)d_in[10];
    const float* fc2b  = (const float*)d_in[11];
    const float* msiW  = (const float*)d_in[12];
    const float* msb1W = (const float*)d_in[14];
    const float* msb2W = (const float*)d_in[16];
    const float* msb3W = (const float*)d_in[18];
    const float* mow   = (const float*)d_in[20];
    const float* mob   = (const float*)d_in[21];
    const float* pxW   = (const float*)d_in[22];
    const float* atW   = (const float*)d_in[24];
    const float* atb   = (const float*)d_in[25];
    const float* coW   = (const float*)d_in[26];
    const float* cob   = (const float*)d_in[27];
    const int* bidx = (const int*)d_in[28];
    const int* nb1  = (const int*)d_in[29];
    const int* nb2  = (const int*)d_in[30];
    const int* nb3  = (const int*)d_in[31];
    float* out = (float*)d_out;

    char* ws = (char*)d_ws;
    float* sm = (float*)ws;
    float* st_g  = sm;          float* st_x  = sm + 128;
    float* st_s  = sm + 256;    float* st_i  = sm + 384;
    float* st_b1 = sm + 512;    float* st_b2 = sm + 640;
    float* st_b3 = sm + 768;    float* st_px = sm + 896;
    float* seg_cnt = sm + 1024;
    float* seg_sum = sm + 1032;
    unsigned* seg_max = (unsigned*)(sm + 1288);
    float* a_ch = sm + 1544;

    size_t off = 65536;
    float* A_sp  = (float*)(ws + off); off += (size_t)n * 4;
    float* A_ms  = (float*)(ws + off); off += (size_t)n * 4;
    float* s_raw = (float*)(ws + off); off += (size_t)n * 4;
    float* craw  = (float*)(ws + off); off += (size_t)n * 128;  // N*32 f32
    float* fused = (float*)(ws + off); off += (size_t)n * 128;
    _Float16* xf16  = (_Float16*)(ws + off); off += (size_t)n * 64;
    _Float16* xi16  = (_Float16*)(ws + off); off += (size_t)n * 64;
    _Float16* t16   = (_Float16*)(ws + off); off += (size_t)n * 64;
    _Float16* cat16 = (_Float16*)(ws + off); off += (size_t)n * 128;
    // packed f16 weights (TDM sources)
    _Float16* wfi  = (_Float16*)(ws + off); off += 27648 * 2;
    _Float16* wf1  = (_Float16*)(ws + off); off += 27648 * 2;
    _Float16* wf2  = (_Float16*)(ws + off); off += 27648 * 2;
    _Float16* wf3  = (_Float16*)(ws + off); off += 27648 * 2;
    _Float16* wfpx = (_Float16*)(ws + off); off += 55296 * 2;
    _Float16* wfat = (_Float16*)(ws + off); off += 1024 * 2;
    _Float16* wfco = (_Float16*)(ws + off); off += 1024 * 2;

    const int n32  = n * 32;
    const int gN32 = (n32 + 255) / 256;
    const int gN   = (n + 255) / 256;
    const float fn = (float)n;
    dim3 b256(256);

    const size_t ldsK27c1 = (size_t)27648 * 2;  // 55296 B
    const size_t ldsK27c2 = (size_t)55296 * 2;  // 110592 B
    const size_t ldsK1c1  = (size_t)1024 * 2;   // 2048 B

    // --- init + weight packing (f32 -> f16, B-operand layout) ---
    zero_small_kernel<<<8, b256, 0, stream>>>(sm, 2048);
    pack_weights_kernel<<<108, b256, 0, stream>>>(msiW,  wfi,  27, 1);
    pack_weights_kernel<<<108, b256, 0, stream>>>(msb1W, wf1,  27, 1);
    pack_weights_kernel<<<108, b256, 0, stream>>>(msb2W, wf2,  27, 1);
    pack_weights_kernel<<<108, b256, 0, stream>>>(msb3W, wf3,  27, 1);
    pack_weights_kernel<<<216, b256, 0, stream>>>(pxW,   wfpx, 27, 2);
    pack_weights_kernel<<<4,   b256, 0, stream>>>(atW,   wfat, 1, 1);
    pack_weights_kernel<<<4,   b256, 0, stream>>>(coW,   wfco, 1, 1);

    // --- spatial gate + channel attention ---
    prep_kernel<<<gN32, b256, 0, stream>>>(x, xf16, cat16, n);
    stage1_stats_kernel<<<160, b256, 0, stream>>>(x, gfeat, wgw, wxw, st_g, st_x, n);
    finalize_stats_kernel<<<1, 32, 0, stream>>>(st_g, 16, fn);
    finalize_stats_kernel<<<1, 32, 0, stream>>>(st_x, 16, fn);
    stage3_kernel<<<160, b256, 0, stream>>>(x, gfeat, wgw, wxw, psiw, st_g, st_x,
                                            s_raw, st_s, n);
    finalize_stats_kernel<<<1, 32, 0, stream>>>(st_s, 1, fn);
    asp_kernel<<<gN, b256, 0, stream>>>(s_raw, st_s, A_sp, n);
    seg_kernel<<<256, b256, 0, stream>>>(x, bidx, seg_cnt, seg_sum, seg_max, n);
    chatt_kernel<<<1, b256, 0, stream>>>(seg_cnt, seg_sum, seg_max,
                                         fc1w, fc1b, fc2w, fc2b, a_ch);

    // --- multi-scale spatial attention (WMMA sparse convs) ---
    conv_wmma_kernel<27, 1><<<512, b256, ldsK27c1, stream>>>(xf16, nb1, wfi,
                                                             craw, st_i, nullptr, n);
    finalize_stats_kernel<<<1, 32, 0, stream>>>(st_i, 32, fn);
    xi_kernel<<<gN32, b256, 0, stream>>>(craw, st_i, xi16, fused, n32);

    conv_wmma_kernel<27, 1><<<512, b256, ldsK27c1, stream>>>(xi16, nb1, wf1,
                                                             craw, st_b1, nullptr, n);
    finalize_stats_kernel<<<1, 32, 0, stream>>>(st_b1, 32, fn);
    fuse_kernel<<<gN32, b256, 0, stream>>>(craw, st_b1, fused, n32);

    conv_wmma_kernel<27, 1><<<512, b256, ldsK27c1, stream>>>(xi16, nb2, wf2,
                                                             craw, st_b2, nullptr, n);
    finalize_stats_kernel<<<1, 32, 0, stream>>>(st_b2, 32, fn);
    fuse_kernel<<<gN32, b256, 0, stream>>>(craw, st_b2, fused, n32);

    conv_wmma_kernel<27, 1><<<512, b256, ldsK27c1, stream>>>(xi16, nb3, wf3,
                                                             craw, st_b3, nullptr, n);
    finalize_stats_kernel<<<1, 32, 0, stream>>>(st_b3, 32, fn);
    fuse_kernel<<<gN32, b256, 0, stream>>>(craw, st_b3, fused, n32);

    ams_kernel<<<gN, b256, 0, stream>>>(fused, mow, mob, A_ms, n);

    // --- triple fusion + 1x1 attn transform (WMMA identity conv) ---
    pattn1pre_kernel<<<gN32, b256, 0, stream>>>(x, A_sp, A_ms, a_ch, bidx, t16, n32);
    conv_wmma_kernel<1, 1><<<512, b256, ldsK1c1, stream>>>(t16, nullptr, wfat,
                                                           craw, nullptr, atb, n);
    cat_hi_kernel<<<gN32, b256, 0, stream>>>(craw, cat16, n32);

    // --- pixel attention conv (Cin=64) ---
    conv_wmma_kernel<27, 2><<<512, b256, ldsK27c2, stream>>>(cat16, nb1, wfpx,
                                                             craw, st_px, nullptr, n);
    finalize_stats_kernel<<<1, 32, 0, stream>>>(st_px, 32, fn);

    // --- residual fusion + 1x1 conv out (WMMA identity conv, writes d_out) ---
    final_pre_kernel<<<gN32, b256, 0, stream>>>(x, craw, st_px, A_sp, t16, n32);
    conv_wmma_kernel<1, 1><<<512, b256, ldsK1c1, stream>>>(t16, nullptr, wfco,
                                                           out, nullptr, cob, n);
}
